// ViG3DEncoder_68710886801881
// MI455X (gfx1250) — matrix operand order repaired and checked by reading, not compiled, not run on hardware
//
#include <hip/hip_runtime.h>

typedef __attribute__((ext_vector_type(16))) _Float16 v16h;
typedef __attribute__((ext_vector_type(8)))  _Float16 v8h;
typedef __attribute__((ext_vector_type(8)))  float    v8f;

#define BN_EPS 1e-5f

// ---------------------------------------------------------------------------
// WMMA helpers (CDNA5 gfx1250, wave32).  D = A(16x32 f16) * B(32x16 f16) + C(f32)
// ---------------------------------------------------------------------------
__device__ __forceinline__ v8f wmma_f16f32(v16h a, v16h b, v8f c) {
  return __builtin_amdgcn_wmma_f32_16x16x32_f16(false, a, false, b, (short)0, c,
                                                false, false);
}

__device__ __forceinline__ v16h cat8(v8h lo, v8h hi) {
  return __builtin_shufflevector(lo, hi, 0, 1, 2, 3, 4, 5, 6, 7,
                                 8, 9, 10, 11, 12, 13, 14, 15);
}

// A fragment from a row-major [16 x ld] f16 LDS tile (ld multiple of 8,
// 16B-aligned base).  lane<16: K={0..7,16..23}; lane>=16: K={8..15,24..31};
// M = lane&15.  Two ds_load_b128 per fragment.
__device__ __forceinline__ v16h load_A_frag(const _Float16* p, int ld) {
  int lane = threadIdx.x & 31;
  int m = lane & 15;
  int kb = (lane < 16) ? 0 : 8;
  v8h lo = *(const v8h*)(p + m * ld + kb);
  v8h hi = *(const v8h*)(p + m * ld + 16 + kb);
  return cat8(lo, hi);
}

// B fragment from a COLUMN-MAJOR [ncols x ldk] f16 LDS tile (Bs[n*ldk + k]).
// lane<16: K={0..7,16..23}; lane>=16: K={8..15,24..31}; N = ncol0 + (lane&15).
// Two ds_load_b128 per fragment.
__device__ __forceinline__ v16h load_B_frag_cm(const _Float16* p, int ldk, int ncol0) {
  int lane = threadIdx.x & 31;
  int n = (lane & 15) + ncol0;
  int kb = (lane < 16) ? 0 : 8;
  v8h lo = *(const v8h*)(p + n * ldk + kb);
  v8h hi = *(const v8h*)(p + n * ldk + 16 + kb);
  return cat8(lo, hi);
}

// ---------------------------------------------------------------------------
// Stem: conv3d 1->48, kernel 4, stride 4, + BN + ReLU.  out: [B][48][16^3]
// ---------------------------------------------------------------------------
__global__ void stem_kernel(const float* __restrict__ x, const float* __restrict__ w,
                            const float* __restrict__ bias, const float* __restrict__ g,
                            const float* __restrict__ beta, float* __restrict__ out) {
  int gid = blockIdx.x * blockDim.x + threadIdx.x;
  const int total = 2 * 48 * 4096;
  if (gid >= total) return;
  int n  = gid & 4095;
  int oc = (gid >> 12) % 48;
  int b  = gid / (48 * 4096);
  int od = n >> 8, oh = (n >> 4) & 15, ow = n & 15;
  float s = 0.f;
#pragma unroll
  for (int i = 0; i < 4; ++i)
#pragma unroll
    for (int j = 0; j < 4; ++j)
#pragma unroll
      for (int k = 0; k < 4; ++k) {
        float xv = x[(size_t)b * 262144 + (size_t)(od * 4 + i) * 4096 +
                     (oh * 4 + j) * 64 + (ow * 4 + k)];
        s += xv * w[((oc * 4 + i) * 4 + j) * 4 + k];
      }
  s += bias[oc];
  s = s * (g[oc] * rsqrtf(1.f + BN_EPS)) + beta[oc];
  out[gid] = fmaxf(s, 0.f);
}

// ---------------------------------------------------------------------------
// Generic GEMM: out[M][N] = epi( BN( W[MxK] @ X[KxN] + bias ) ), per batch z.
// epi: 0 = none (+optional f16 copy), 1 = tanh-GELU, 2 = +residual.
// grid (N/64, M/16, B), block 128 (4 waves, one 16x16 tile per wave).
// As row-major [16][32]; Bs column-major [64 cols][32 k].
// ---------------------------------------------------------------------------
__global__ __launch_bounds__(128) void gemm_bn_kernel(
    const float* __restrict__ W, const float* __restrict__ X,
    const float* __restrict__ bias, const float* __restrict__ bng,
    const float* __restrict__ bnb, const float* __restrict__ resid,
    float* __restrict__ out, _Float16* __restrict__ out16,
    int M, int Kd, int N, int epi) {
  __shared__ __align__(16) _Float16 As[16 * 32];
  __shared__ __align__(16) _Float16 Bs[64 * 32];  // column-major: Bs[n*32 + k]
  int b = blockIdx.z;
  int m0 = blockIdx.y * 16;
  int n0 = blockIdx.x * 64;
  const float* Xb = X + (size_t)b * Kd * N;
  int tid = threadIdx.x;
  int wave = tid >> 5, lane = tid & 31;
  v8f acc = {};
  int kchunks = (Kd + 31) >> 5;
  for (int kc = 0; kc < kchunks; ++kc) {
    int k0 = kc << 5;
    bool full = (k0 + 32 <= Kd);
    if (full) {
      for (int i = tid; i < 16 * 32; i += 128) {
        int m = i >> 5, k = i & 31;
        As[i] = (_Float16)W[(m0 + m) * Kd + k0 + k];
      }
      for (int i = tid; i < 32 * 64; i += 128) {
        int k = i >> 6, n = i & 63;  // consecutive n -> coalesced global read
        Bs[n * 32 + k] = (_Float16)Xb[(size_t)(k0 + k) * N + n0 + n];
      }
    } else {
      for (int i = tid; i < 16 * 32; i += 128) {
        int m = i >> 5, k = i & 31;
        As[i] = (_Float16)((k0 + k < Kd) ? W[(m0 + m) * Kd + k0 + k] : 0.f);
      }
      for (int i = tid; i < 32 * 64; i += 128) {
        int k = i >> 6, n = i & 63;
        Bs[n * 32 + k] =
            (_Float16)((k0 + k < Kd) ? Xb[(size_t)(k0 + k) * N + n0 + n] : 0.f);
      }
    }
    if (k0 + 32 < Kd)  // prefetch next B panel (global_prefetch_b8)
      __builtin_prefetch(Xb + (size_t)(k0 + 32) * N + n0 + (tid & 63), 0, 3);
    __syncthreads();
    v16h af = load_A_frag(As, 32);
    v16h bf = load_B_frag_cm(Bs, 32, wave * 16);
    acc = wmma_f16f32(af, bf, acc);
    __syncthreads();
  }
  int col = n0 + wave * 16 + (lane & 15);
  int rbase = (lane < 16) ? 0 : 8;
#pragma unroll
  for (int r = 0; r < 8; ++r) {
    int m = m0 + rbase + r;
    float y = acc[r] + bias[m];
    y = y * (bng[m] * rsqrtf(1.f + BN_EPS)) + bnb[m];
    if (epi == 1) {
      float c = y + 0.044715f * y * y * y;
      y = 0.5f * y * (1.f + tanhf(0.7978845608028654f * c));
    } else if (epi == 2) {
      y += resid[(size_t)b * M * N + (size_t)m * N + col];
    }
    size_t o = (size_t)b * M * N + (size_t)m * N + col;
    out[o] = y;
    if (out16) out16[o] = (_Float16)y;
  }
}

// ---------------------------------------------------------------------------
// sq[b][n] = sum_c h[b][c][n]^2
// ---------------------------------------------------------------------------
__global__ void sqnorm_kernel(const float* __restrict__ h, float* __restrict__ sq,
                              int C, int N) {
  int n = blockIdx.x * blockDim.x + threadIdx.x;
  int b = blockIdx.y;
  if (n >= N) return;
  const float* hb = h + (size_t)b * C * N;
  float s = 0.f;
  for (int c = 0; c < C; ++c) { float v = hb[(size_t)c * N + n]; s += v * v; }
  sq[(size_t)b * N + n] = s;
}

// ---------------------------------------------------------------------------
// kNN top-9: one wave per 32 rows.  Gram tiles via WMMA, per-lane top-9 heap.
// h16 layout [B][C][N].  idx out: [B][N][9].
// grid (N/32, B), block 32.
// At row-major [32 rows][CPAD]; Bs column-major [16 cols][CPAD].
// ---------------------------------------------------------------------------
#define CPAD 192
__global__ __launch_bounds__(32) void knn_topk_kernel(
    const _Float16* __restrict__ h16, const float* __restrict__ sq,
    int* __restrict__ idx, int C, int N) {
  __shared__ __align__(16) _Float16 At[32 * CPAD];
  __shared__ __align__(16) _Float16 Bs[16 * CPAD];  // Bs[n*CPAD + k]
  __shared__ __align__(16) float    tile[32 * 16];
  int b = blockIdx.y;
  int n0 = blockIdx.x * 32;
  int lane = threadIdx.x;
  const _Float16* hb = h16 + (size_t)b * C * N;
  const float* sqb = sq + (size_t)b * N;

  // preload the 32-row A block (transposed gather from [C][N]); no predication
  for (int c = 0; c < C; ++c)
    At[lane * CPAD + c] = hb[(size_t)c * N + n0 + lane];
  for (int c = C; c < CPAD; ++c)
    At[lane * CPAD + c] = (_Float16)0.f;

  float bd[9]; int bi[9];
#pragma unroll
  for (int j = 0; j < 9; ++j) { bd[j] = 3.0e38f; bi[j] = 0; }
  float sqn = sqb[n0 + lane];
  int kchunks = (C + 31) >> 5;

  for (int m0 = 0; m0 < N; m0 += 16) {
    // load column tile, coalesced on n, scattered into column-major LDS
    for (int i = lane; i < C * 16; i += 32) {
      int k = i >> 4, n = i & 15;
      Bs[n * CPAD + k] = hb[(size_t)k * N + m0 + n];
    }
    for (int i = lane + C * 16; i < CPAD * 16; i += 32) {
      int k = i >> 4, n = i & 15;
      Bs[n * CPAD + k] = (_Float16)0.f;
    }
    __syncthreads();
#pragma unroll 1
    for (int t = 0; t < 2; ++t) {
      v8f acc = {};
      for (int kc = 0; kc < kchunks; ++kc) {
        v16h af = load_A_frag(At + (16 * t) * CPAD + kc * 32, CPAD);
        v16h bf = load_B_frag_cm(Bs + kc * 32, CPAD, 0);
        acc = wmma_f16f32(af, bf, acc);
      }
      int row = 16 * t + ((lane < 16) ? 0 : 8);
      int colr = lane & 15;
#pragma unroll
      for (int r = 0; r < 8; ++r) tile[(row + r) * 16 + colr] = acc[r];
    }
    __syncthreads();
    const float* trow = tile + lane * 16;
#pragma unroll
    for (int j2 = 0; j2 < 16; ++j2) {
      int m = m0 + j2;
      float d2 = sqn + sqb[m] - 2.f * trow[j2];
      if (d2 < bd[8]) {
        bd[8] = d2; bi[8] = m;
#pragma unroll
        for (int q = 8; q > 0; --q) {
          if (bd[q] < bd[q - 1]) {
            float td = bd[q]; bd[q] = bd[q - 1]; bd[q - 1] = td;
            int ti = bi[q]; bi[q] = bi[q - 1]; bi[q - 1] = ti;
          }
        }
      }
    }
    __syncthreads();
  }
  int n = n0 + lane;
#pragma unroll
  for (int j = 0; j < 9; ++j) idx[((size_t)b * N + n) * 9 + j] = bi[j];
}

// ---------------------------------------------------------------------------
// cat[b][0:C][n] = h ; cat[b][C:2C][n] = max_k ( h[c][idx[n][k]] - h[c][n] )
// ---------------------------------------------------------------------------
__global__ void maxrel_cat_kernel(const float* __restrict__ h1,
                                  const int* __restrict__ idx,
                                  float* __restrict__ cat, int C, int N) {
  int gid = blockIdx.x * blockDim.x + threadIdx.x;
  int total = 2 * C * N;
  if (gid >= total) return;
  int n = gid % N;
  int c = (gid / N) % C;
  int b = gid / (N * C);
  const float* hb = h1 + (size_t)b * C * N;
  float hv = hb[(size_t)c * N + n];
  const int* ib = idx + ((size_t)b * N + n) * 9;
  float mr = -3.0e38f;
#pragma unroll
  for (int k = 0; k < 9; ++k) mr = fmaxf(mr, hb[(size_t)c * N + ib[k]] - hv);
  float* cb = cat + (size_t)b * 2 * C * N;
  cb[(size_t)c * N + n] = hv;
  cb[(size_t)(C + c) * N + n] = mr;
}

// ---------------------------------------------------------------------------
// ReLU -> d_out slice and in-place workspace copy
// ---------------------------------------------------------------------------
__global__ void relu_out_kernel(const float* __restrict__ y, float* __restrict__ dst,
                                float* __restrict__ wsdst, int n) {
  int i = blockIdx.x * blockDim.x + threadIdx.x;
  if (i >= n) return;
  float v = fmaxf(y[i], 0.f);
  dst[i] = v;
  wsdst[i] = v;
}

// ---------------------------------------------------------------------------
// Downsample: conv3d kernel 2 stride 2, Cin->Cout, + BN + ReLU.
// in [B][Cin][(2S)^3], out [B][Cout][S^3]
// ---------------------------------------------------------------------------
__global__ void down_kernel(const float* __restrict__ in, const float* __restrict__ w,
                            const float* __restrict__ bias, const float* __restrict__ g,
                            const float* __restrict__ beta, float* __restrict__ out,
                            int Cin, int Cout, int S) {
  int gid = blockIdx.x * blockDim.x + threadIdx.x;
  int NS = S * S * S;
  int total = 2 * Cout * NS;
  if (gid >= total) return;
  int n = gid % NS;
  int oc = (gid / NS) % Cout;
  int b = gid / (NS * Cout);
  int od = n / (S * S), oh = (n / S) % S, ow = n % S;
  int Sin = 2 * S;
  float s = 0.f;
  for (int ic = 0; ic < Cin; ++ic) {
    const float* ibase = in + ((size_t)b * Cin + ic) * Sin * Sin * Sin;
    const float* wbase = w + ((size_t)oc * Cin + ic) * 8;
#pragma unroll
    for (int i = 0; i < 2; ++i)
#pragma unroll
      for (int j = 0; j < 2; ++j)
#pragma unroll
        for (int k = 0; k < 2; ++k)
          s += ibase[(size_t)(2 * od + i) * Sin * Sin + (2 * oh + j) * Sin +
                     (2 * ow + k)] * wbase[(i * 2 + j) * 2 + k];
  }
  s += bias[oc];
  s = s * (g[oc] * rsqrtf(1.f + BN_EPS)) + beta[oc];
  out[gid] = fmaxf(s, 0.f);
}

// ---------------------------------------------------------------------------
// Host launch.  Flattened input order (sorted pytree keys):
//  0-3 down0 {b,beta,g,w}, 4-7 down1, 8.. stages (12 leaves each, sorted:
//  bn1_b,bn1_g,bn2_b,bn2_g,bn_gc_b,bn_gc_g,fc1_b,fc1_w,fc2_b,fc2_w,gc_b,gc_w),
//  44 stem_b, 45 stem_bn_b, 46 stem_bn_g, 47 stem_w, 48 x.
// ---------------------------------------------------------------------------
extern "C" void kernel_launch(void* const* d_in, const int* in_sizes, int n_in,
                              void* d_out, int out_size, void* d_ws, size_t ws_size,
                              hipStream_t stream) {
  (void)in_sizes; (void)n_in; (void)out_size; (void)ws_size;
  const float* x       = (const float*)d_in[48];
  const float* stem_b  = (const float*)d_in[44];
  const float* stem_bnb= (const float*)d_in[45];
  const float* stem_bng= (const float*)d_in[46];
  const float* stem_w  = (const float*)d_in[47];

  // workspace partition (floats)
  float* hin  = (float*)d_ws;          // 393216 : block input / shortcut
  float* h1v  = hin  + 393216;         // 393216 : fc1 output
  float* catv = h1v  + 393216;         // 786432 : concat [2C][N]
  float* gv   = catv + 786432;         // 786432 : gc output
  float* yv   = gv   + 786432;         // 393216 : fc2 + residual
  float* sqv  = yv   + 393216;         // 8192
  int*   idxv = (int*)(sqv + 8192);    // 73728
  _Float16* h16 = (_Float16*)(idxv + 73728);  // 393216 halves

  stem_kernel<<<dim3((2 * 48 * 4096 + 255) / 256), 256, 0, stream>>>(
      x, stem_w, stem_b, stem_bng, stem_bnb, hin);

  const int Cs[3] = {48, 96, 192};
  const int Ns[3] = {4096, 512, 64};
  const int pb[3] = {8, 20, 32};
  const size_t outoff[3] = {0, 393216, 491520};
  float* out = (float*)d_out;

  for (int s = 0; s < 3; ++s) {
    int C = Cs[s], N = Ns[s];
    const float* bn1_b  = (const float*)d_in[pb[s] + 0];
    const float* bn1_g  = (const float*)d_in[pb[s] + 1];
    const float* bn2_b  = (const float*)d_in[pb[s] + 2];
    const float* bn2_g  = (const float*)d_in[pb[s] + 3];
    const float* bngc_b = (const float*)d_in[pb[s] + 4];
    const float* bngc_g = (const float*)d_in[pb[s] + 5];
    const float* fc1_b  = (const float*)d_in[pb[s] + 6];
    const float* fc1_w  = (const float*)d_in[pb[s] + 7];
    const float* fc2_b  = (const float*)d_in[pb[s] + 8];
    const float* fc2_w  = (const float*)d_in[pb[s] + 9];
    const float* gc_b   = (const float*)d_in[pb[s] + 10];
    const float* gc_w   = (const float*)d_in[pb[s] + 11];

    // fc1 + BN (f32 out + f16 copy for the Gram matmul)
    gemm_bn_kernel<<<dim3(N / 64, C / 16, 2), 128, 0, stream>>>(
        fc1_w, hin, fc1_b, bn1_g, bn1_b, nullptr, h1v, h16, C, C, N, 0);
    // squared norms
    sqnorm_kernel<<<dim3((N + 255) / 256, 2), 256, 0, stream>>>(h1v, sqv, C, N);
    // dynamic kNN (WMMA Gram + per-lane top-9)
    knn_topk_kernel<<<dim3(N / 32, 2), 32, 0, stream>>>(h16, sqv, idxv, C, N);
    // gather + max-relative + concat
    maxrel_cat_kernel<<<dim3((2 * C * N + 255) / 256), 256, 0, stream>>>(
        h1v, idxv, catv, C, N);
    // graph conv fc + BN + GELU
    gemm_bn_kernel<<<dim3(N / 64, (2 * C) / 16, 2), 128, 0, stream>>>(
        gc_w, catv, gc_b, bngc_g, bngc_b, nullptr, gv, nullptr, 2 * C, 2 * C, N, 1);
    // fc2 + BN + residual
    gemm_bn_kernel<<<dim3(N / 64, C / 16, 2), 128, 0, stream>>>(
        fc2_w, gv, fc2_b, bn2_g, bn2_b, hin, yv, nullptr, C, 2 * C, N, 2);
    // stage ReLU -> feats slice (+ keep relu'd copy for downsample)
    relu_out_kernel<<<dim3((2 * C * N + 255) / 256), 256, 0, stream>>>(
        yv, out + outoff[s], yv, 2 * C * N);
    // downsample to next stage
    if (s < 2) {
      const float* dw    = (const float*)d_in[s * 4 + 3];
      const float* db    = (const float*)d_in[s * 4 + 0];
      const float* dbeta = (const float*)d_in[s * 4 + 1];
      const float* dg    = (const float*)d_in[s * 4 + 2];
      int Co = Cs[s + 1];
      int So = (s == 0) ? 8 : 4;
      down_kernel<<<dim3((2 * Co * So * So * So + 255) / 256), 256, 0, stream>>>(
          yv, dw, db, dg, dbeta, hin, C, Co, So);
    }
  }
}
